// CAWN2_36593121362101
// MI455X (gfx1250) — compile-verified, matrix operand order checked by reading, not run on hardware
//
#include <hip/hip_runtime.h>

// CDNA5 / gfx1250, wave32 WMMA types
typedef __attribute__((ext_vector_type(16))) __bf16 v16bf;
typedef __attribute__((ext_vector_type(8)))  float  v8f;

#define FEAT   128
#define MODEL  384
#define BROWS  131072
#define JT     24      // 384/16 output-column tiles
#define KC     12      // 384/32 K chunks

__device__ __forceinline__ unsigned short f2bf(float f) {
    union { float f; unsigned u; } v; v.f = f;
    unsigned u = v.u;
    u += 0x7fffu + ((u >> 16) & 1u);     // round-to-nearest-even
    return (unsigned short)(u >> 16);
}

__device__ __forceinline__ float fast_tanh(float x) {
#if __has_builtin(__builtin_amdgcn_tanhf)
    return __builtin_amdgcn_tanhf(x);    // v_tanh_f32 (CDNA5 TRANS op)
#else
    float t = __expf(-2.f * fabsf(x));
    float r = (1.f - t) / (1.f + t);
    return copysignf(r, x);
#endif
}

__device__ __forceinline__ float fast_sigmoid(float x) {
    return __builtin_fmaf(0.5f, fast_tanh(0.5f * x), 0.5f);
}

union Frag { v16bf v; unsigned short u[16]; };

// ---------------------------------------------------------------------------
// Phase 0: swizzle w_ih (f32 [1536,384]) rows for gates {i,g,o} into bf16
// WMMA B-fragments, jt-major so one jt's 3 gate chunks are contiguous (36 KB).
// Fragment (jt, g, kc): 512 bf16; lane L holds column n = jt*16 + L%16,
// K = kc*32 + {half*8..+8} U {16+half*8..+8}, half = L/16.
// ---------------------------------------------------------------------------
__global__ void prep_w(const float* __restrict__ w_ih,
                       unsigned short* __restrict__ Bw) {
    int t = blockIdx.x * blockDim.x + threadIdx.x;
    if (t >= 3 * JT * KC * 32) return;
    int lane = t & 31;
    int kc   = (t >> 5) % KC;
    int g    = (t / (32 * KC)) % 3;
    int jt   =  t / (32 * KC * 3);
    const int goff[3] = {0, 2 * MODEL, 3 * MODEL};   // i, g, o row offsets
    int half = lane >> 4;
    int n    = lane & 15;
    const float* wr = w_ih + (long)(goff[g] + jt * 16 + n) * MODEL;
    int k0 = kc * 32 + half * 8;

    unsigned short tmp[16];
#pragma unroll
    for (int e = 0; e < 8; ++e) tmp[e]     = f2bf(wr[k0 + e]);
#pragma unroll
    for (int e = 0; e < 8; ++e) tmp[8 + e] = f2bf(wr[k0 + 16 + e]);

    uint4* dst = (uint4*)(Bw + (((jt * 3 + g) * KC + kc) * 512 + lane * 16));
    dst[0] = ((const uint4*)tmp)[0];
    dst[1] = ((const uint4*)tmp)[1];
}

// ---------------------------------------------------------------------------
// Phase 1: fused gather + time-encode + bf16 GEMM (WMMA) + LSTM epilogue.
// Grid: BROWS/128 blocks of 256 threads (8 waves, 16 rows per wave).
// ---------------------------------------------------------------------------
__global__ void __launch_bounds__(256)
cawn_lstm(const int* __restrict__ src_idx, const int* __restrict__ tgt_idx,
          const float* __restrict__ cut_time, const int* __restrict__ e_idx,
          const float* __restrict__ node_feat, const float* __restrict__ edge_feat,
          const float* __restrict__ basis_freq, const float* __restrict__ phase,
          const unsigned short* __restrict__ Bw,
          const float* __restrict__ b_ih, const float* __restrict__ b_hh,
          float* __restrict__ out) {
    __shared__ __align__(32) unsigned short ldsB[3 * KC * 512];   // 36 KB/jt

    int tid  = threadIdx.x;
    int wave = tid >> 5;
    int lane = tid & 31;
    int half = lane >> 4;         // A: K-half select; C/D: M high-half select
    int l16  = lane & 15;         // A: row M; C/D: col N
    int r0   = blockIdx.x * 128 + wave * 16;
    int row  = r0 + l16;

    // --- build the 12 bf16 A-fragments for this wave's 16 agg rows ---------
    Frag a[KC];
    {
        const float* ps = node_feat + (long)src_idx[row] * FEAT;
        const float* pt = node_feat + (long)tgt_idx[row] * FEAT;
#pragma unroll
        for (int kc = 0; kc < 4; ++kc) {               // hidden embeddings
            int k0 = kc * 32 + half * 8;
#pragma unroll
            for (int e = 0; e < 8; ++e) a[kc].u[e]     = f2bf(ps[k0 + e]      + pt[k0 + e]);
#pragma unroll
            for (int e = 0; e < 8; ++e) a[kc].u[8 + e] = f2bf(ps[k0 + 16 + e] + pt[k0 + 16 + e]);
        }
        float ct = cut_time[row];
#pragma unroll
        for (int kc = 4; kc < 8; ++kc) {               // time encode: cos(t*w + p)
            int k0 = kc * 32 + half * 8 - FEAT;
#pragma unroll
            for (int e = 0; e < 8; ++e)
                a[kc].u[e]     = f2bf(__cosf(ct * basis_freq[k0 + e]      + phase[k0 + e]));
#pragma unroll
            for (int e = 0; e < 8; ++e)
                a[kc].u[8 + e] = f2bf(__cosf(ct * basis_freq[k0 + 16 + e] + phase[k0 + 16 + e]));
        }
        const float* pe = edge_feat + (long)e_idx[row] * FEAT;
#pragma unroll
        for (int kc = 8; kc < 12; ++kc) {              // edge features
            int k0 = kc * 32 + half * 8 - 2 * FEAT;
#pragma unroll
            for (int e = 0; e < 8; ++e) a[kc].u[e]     = f2bf(pe[k0 + e]);
#pragma unroll
            for (int e = 0; e < 8; ++e) a[kc].u[8 + e] = f2bf(pe[k0 + 16 + e]);
        }
    }

    float* outh = out;
    float* outc = out + (long)BROWS * MODEL;

    for (int jt = 0; jt < JT; ++jt) {
        // ---- stage all 3 gate chunks for this jt: 2304 uint4 = 36 KB ------
        __syncthreads();
        {
            const uint4* gsrc = (const uint4*)(Bw + (long)jt * (3 * KC * 512));
            uint4*       gdst = (uint4*)ldsB;
#pragma unroll
            for (int i = 0; i < 9; ++i)
                gdst[tid + 256 * i] = gsrc[tid + 256 * i];
        }
        __syncthreads();

        v8f acc[3];
#pragma unroll
        for (int g = 0; g < 3; ++g) {
            v8f c = {0.f, 0.f, 0.f, 0.f, 0.f, 0.f, 0.f, 0.f};
#pragma unroll
            for (int kc = 0; kc < KC; ++kc) {
                v16bf bf = *(const v16bf*)(&ldsB[(g * KC + kc) * 512 + lane * 16]);
                c = __builtin_amdgcn_wmma_f32_16x16x32_bf16(
                        false, a[kc].v, false, bf, (short)0, c, false, false);
            }
            acc[g] = c;
        }

        // --- LSTM epilogue: c = sig(i)*tanh(g); h = sig(o)*tanh(c) ----------
        int col = jt * 16 + l16;
        float bi = b_ih[col]             + b_hh[col];
        float bg = b_ih[2 * MODEL + col] + b_hh[2 * MODEL + col];
        float bo = b_ih[3 * MODEL + col] + b_hh[3 * MODEL + col];
#pragma unroll
        for (int v = 0; v < 8; ++v) {
            int  rr = r0 + v + 8 * half;               // C/D layout: M = v + 8*(lane/16)
            float iv = acc[0][v] + bi;
            float gv = acc[1][v] + bg;
            float ov = acc[2][v] + bo;
            float cc = fast_sigmoid(iv) * fast_tanh(gv);
            float hh = fast_sigmoid(ov) * fast_tanh(cc);
            outh[(long)rr * MODEL + col] = hh;
            outc[(long)rr * MODEL + col] = cc;
        }
    }
}

// ---------------------------------------------------------------------------
extern "C" void kernel_launch(void* const* d_in, const int* in_sizes, int n_in,
                              void* d_out, int out_size, void* d_ws, size_t ws_size,
                              hipStream_t stream) {
    const int*   src_idx    = (const int*)  d_in[0];
    const int*   tgt_idx    = (const int*)  d_in[1];
    const float* cut_time   = (const float*)d_in[2];
    const int*   e_idx      = (const int*)  d_in[3];
    const float* node_feat  = (const float*)d_in[4];
    const float* edge_feat  = (const float*)d_in[5];
    const float* basis_freq = (const float*)d_in[6];
    const float* phase      = (const float*)d_in[7];
    const float* w_ih       = (const float*)d_in[8];
    // d_in[9] = w_hh: unused (h0 == 0 -> h0 @ w_hh.T == 0)
    const float* b_ih       = (const float*)d_in[10];
    const float* b_hh       = (const float*)d_in[11];

    unsigned short* Bw = (unsigned short*)d_ws;  // 3*24*12*512*2 = 884,736 bytes

    prep_w<<<(3 * JT * KC * 32 + 255) / 256, 256, 0, stream>>>(w_ih, Bw);
    cawn_lstm<<<BROWS / 128, 256, 0, stream>>>(
        src_idx, tgt_idx, cut_time, e_idx, node_feat, edge_feat,
        basis_freq, phase, Bw, b_ih, b_hh, (float*)d_out);
}
